// SupConLoss_21663815041494
// MI455X (gfx1250) — matrix operand order repaired and checked by reading, not compiled, not run on hardware
//
#include <hip/hip_runtime.h>
#include <hip/hip_bf16.h>

// ---------------------------------------------------------------------------
// SupConLoss fused kernels for MI455X (gfx1250, wave32, WMMA + TDM).
//
// sim = (Fn Fn^T)/T over N=8192, D=512: 68.7 GFLOP GEMM. bf16 features are
// 8 MB -> L2 resident -> compute-bound:
//  * never materialize the 256 MB sim matrix (fuse softmax/mask stats),
//  * share B tiles across waves through LDS (64-row strips -> ~1 GB L2 reads),
//  * stage A strip and double-buffered B tiles with the Tensor Data Mover
//    (tensor_load_to_lds, TENSORcnt) so DMA of tile it+1 overlaps WMMA of
//    tile it and no VALU/vmem slots are spent on staging copies,
//  * WMMA operands flow LDS->VGPR each iteration (no spills, round-1 fix).
// ---------------------------------------------------------------------------

#define B_SZ   4096
#define VIEWS  2
#define D_SZ   512
#define N_SZ   (B_SZ * VIEWS)        // 8192
#define INV_T  (1.0f / 0.07f)
#define LDS_STRIDE 520               // 512 + 8 bf16 (TDM pad: 4 DW per 256 DW)

#define MB      64                   // anchor rows per block
#define CT      32                   // columns staged per iteration
#define SPLITS  4                    // column-split for parallelism
#define COLS_PER_SPLIT (N_SZ / SPLITS)   // 2048
#define ITERS   (COLS_PER_SPLIT / CT)    // 64

typedef __attribute__((ext_vector_type(16))) __bf16 v16bf;
typedef __attribute__((ext_vector_type(8)))  float  v8f;
typedef __attribute__((ext_vector_type(4)))  unsigned int v4u;
typedef __attribute__((ext_vector_type(8)))  int    v8i;
typedef __attribute__((ext_vector_type(4)))  int    v4i;

union ABFrag {
    v16bf v;
    uint4 q[2];
};

// Issue one TDM 2D tile load: nrows x 512 bf16, row stride 512 (contiguous in
// memory), into LDS at lds_off with a 4-DWORD pad after every 256 DWORDs
// (one row) -> effective LDS row stride = LDS_STRIDE bf16. D# per ISA ch.8.
__device__ __forceinline__ void tdm_load_tile(const unsigned short* gsrc,
                                              unsigned lds_off, int nrows) {
    const unsigned long long ga = (unsigned long long)(uintptr_t)gsrc;
    v4u g0;
    g0.x = 0x1u;                                        // count=1 (valid), user D#
    g0.y = lds_off;                                     // lds_addr (bytes)
    g0.z = (unsigned)(ga & 0xFFFFFFFFu);                // global_addr[31:0]
    g0.w = (unsigned)((ga >> 32) & 0x1FFFFFFu)          // global_addr[56:32]
         | (2u << 30);                                  // type = 2 ("image")
    v8i g1;
    g1[0] = (1 << 16)        // data_size = 2 bytes
          | (1 << 20)        // pad_enable
          | (7 << 22)        // pad_interval code 7 -> every 256 DWORDs (1 row)
          | (3 << 25);       // pad_amount  code 3 -> 4 DWORDs (16 B)
    g1[1] = (int)(512u << 16);          // tensor_dim0 = 512 (low16); bar addr 0
    g1[2] = (int)(0x2000u << 16);       // dim0 hi16=0 | tensor_dim1 = 8192 low16
    g1[3] = (int)(512u << 16);          // dim1 hi16=0 | tile_dim0 = 512
    g1[4] = nrows;                      // tile_dim1 = nrows; tile_dim2 = 0
    g1[5] = 512;                        // tensor_dim0_stride = 512 (low 32)
    g1[6] = 0;                          // stride hi | tensor_dim1_stride lo = 0
    g1[7] = 0;
    const v4i z4 = {0, 0, 0, 0};        // groups 2/3 unused (2D tile)
#if defined(__clang_major__) && __clang_major__ >= 23
    const v8i z8 = {0, 0, 0, 0, 0, 0, 0, 0};
    __builtin_amdgcn_tensor_load_to_lds(g0, g1, z4, z4, z8, 0);
#else
    __builtin_amdgcn_tensor_load_to_lds(g0, g1, z4, z4, 0);
#endif
}

// --- Kernel 1: L2-normalize each (b,v) row, convert to bf16 (RNE), store
// --- view-major (row n = v*B + b). One wave per row.
__global__ void normalize_bf16(const float* __restrict__ feats,
                               unsigned short* __restrict__ out) {
    const int wave = threadIdx.x >> 5;
    const int lane = threadIdx.x & 31;
    const int n = blockIdx.x * 8 + wave;
    const int b = n & (B_SZ - 1);
    const int v = n >> 12;
    const float* src = feats + ((size_t)b * VIEWS + v) * D_SZ;

    float vals[16];
    float ss = 0.f;
#pragma unroll
    for (int t = 0; t < 16; ++t) {
        float f = src[t * 32 + lane];
        vals[t] = f;
        ss += f * f;
    }
#pragma unroll
    for (int m = 16; m >= 1; m >>= 1) ss += __shfl_xor(ss, m, 32);

    const float scale = 1.f / fmaxf(sqrtf(ss), 1e-8f);
    unsigned short* dst = out + (size_t)n * D_SZ;
#pragma unroll
    for (int t = 0; t < 16; ++t) {
        float f = vals[t] * scale;
        unsigned int u = __float_as_uint(f);
        unsigned int r = u + 0x7FFFu + ((u >> 16) & 1u);   // RNE
        dst[t * 32 + lane] = (unsigned short)(r >> 16);
    }
}

// --- Kernel 2: fused GEMM + masked-softmax statistics.
// Grid: (N/MB)*SPLITS = 512 blocks of 256 threads (8 waves).
// blockIdx.x = strip*SPLITS + split. Block owns MB=64 anchor rows (A strip,
// TDM-loaded once) and sweeps its 2048-column range in CT=32-column tiles
// (TDM double-buffered). Waves: rowGroup = w&3, colGroup = w>>2.
// Per-row stats: S = sum_{j!=i} exp(clip(sim)), P = sum_{pos} clip(sim),
// C = #pos over the swept columns.
__global__ void __launch_bounds__(256)
supcon_main(const unsigned short* __restrict__ Fbf,
            const int* __restrict__ labels,
            float* __restrict__ gstats) {
    __shared__ __align__(16) unsigned short Atile[MB * LDS_STRIDE];
    __shared__ __align__(16) unsigned short Btile[2][CT * LDS_STRIDE];
    __shared__ float ldsS[MB], ldsP[MB], ldsC[MB];

    const int tid  = threadIdx.x;
    const int wave = tid >> 5;
    const int lane = tid & 31;
    const int half = lane >> 4;
    const int l16  = lane & 15;
    const int rowGroup = wave & 3;
    const int colGroup = wave >> 2;

    const int strip = blockIdx.x >> 2;          // /SPLITS
    const int split = blockIdx.x & (SPLITS - 1);
    const int rowBase  = strip * MB;
    const int colBase0 = split * COLS_PER_SPLIT;

    // TDM prologue: A strip (64 rows) + first B tile (32 cols), in-order ops.
    if (wave == 0) {
        tdm_load_tile(Fbf + (size_t)rowBase * D_SZ,
                      (unsigned)(uintptr_t)&Atile[0], MB);
        tdm_load_tile(Fbf + (size_t)colBase0 * D_SZ,
                      (unsigned)(uintptr_t)&Btile[0][0], CT);
    }
    if (tid < MB) { ldsS[tid] = 0.f; ldsP[tid] = 0.f; ldsC[tid] = 0.f; }

    // Anchor labels for this lane's 8 accumulator rows.
    int labI[8];
#pragma unroll
    for (int r = 0; r < 8; ++r)
        labI[r] = labels[(rowBase + rowGroup * 16 + r + 8 * half) & (B_SZ - 1)];

    float accS[8], accP[8], accC[8];
#pragma unroll
    for (int r = 0; r < 8; ++r) { accS[r] = 0.f; accP[r] = 0.f; accC[r] = 0.f; }

    for (int it = 0; it < ITERS; ++it) {
        const int cur = it & 1;
        const int colBase = colBase0 + it * CT;

        // Wait for the in-flight TDM transfer(s) targeting Btile[cur] (and,
        // on it==0, the A strip), then rendezvous the workgroup.
        if (wave == 0) __builtin_amdgcn_s_wait_tensorcnt(0);
        __syncthreads();

        // Kick off the DMA of the next tile into the other buffer; it runs
        // concurrently with this iteration's WMMA stream.
        if (wave == 0 && (it + 1) < ITERS) {
            tdm_load_tile(Fbf + (size_t)(colBase + CT) * D_SZ,
                          (unsigned)(uintptr_t)&Btile[cur ^ 1][0], CT);
        }

        // Opaque zero offset: keeps A-fragment LDS reads inside the loop
        // (prevents the 128-VGPR hoist + scratch spill seen in round 1).
        int az = it;
        asm("" : "+v"(az));
        az -= it;

        const unsigned short* ap =
            Atile + (size_t)(rowGroup * 16 + l16) * LDS_STRIDE + half * 8 + az;
        const unsigned short* bp =
            Btile[cur] + (size_t)(colGroup * 16 + l16) * LDS_STRIDE + half * 8;

        const int col = colBase + colGroup * 16 + l16;     // this lane's column
        const int labJ = labels[col & (B_SZ - 1)];

        v8f acc = {};
#pragma unroll
        for (int k0 = 0; k0 < 16; ++k0) {
            ABFrag a, b;
            // A frag (16-bit 16x32 layout): two 16B chunks, K = k0*32 + h*8
            // and K = k0*32 + 16 + h*8.
            a.q[0] = *(const uint4*)(ap + k0 * 32);
            a.q[1] = *(const uint4*)(ap + k0 * 32 + 16);
            // B frag (32x16): two 16B chunks, K = k0*32 + h*8
            // and K = k0*32 + 16 + h*8.
            b.q[0] = *(const uint4*)(bp + k0 * 32);
            b.q[1] = *(const uint4*)(bp + k0 * 32 + 16);
            acc = __builtin_amdgcn_wmma_f32_16x16x32_bf16(
                      false, a.v, false, b.v, (short)0, acc, false, false);
        }

        // Epilogue: VGPR r, half h holds element (M = r+8h, N = l16).
#pragma unroll
        for (int r = 0; r < 8; ++r) {
            const int i = rowBase + rowGroup * 16 + r + 8 * half;
            float s = acc[r] * INV_T;
            s = fminf(fmaxf(s, -20.f), 20.f);
            const bool diag = (i == col);
            accS[r] += diag ? 0.f : __expf(s);
            const bool pos = (!diag) && (labJ == labI[r]);
            accP[r] += pos ? s : 0.f;
            accC[r] += pos ? 1.f : 0.f;
        }
        __syncthreads();   // all waves done reading Btile[cur] before reuse
    }

    // Reduce each row's stats across the 16 lanes of the half-wave.
#pragma unroll
    for (int r = 0; r < 8; ++r) {
#pragma unroll
        for (int m = 1; m < 16; m <<= 1) {
            accS[r] += __shfl_xor(accS[r], m, 32);
            accP[r] += __shfl_xor(accP[r], m, 32);
            accC[r] += __shfl_xor(accC[r], m, 32);
        }
    }
    if (l16 == 0) {
#pragma unroll
        for (int r = 0; r < 8; ++r) {
            const int row = rowGroup * 16 + r + 8 * half;
            atomicAdd(&ldsS[row], accS[r]);    // ds_add_f32 (2 colGroups/row)
            atomicAdd(&ldsP[row], accP[r]);
            atomicAdd(&ldsC[row], accC[r]);
        }
    }
    __syncthreads();

    if (tid < MB) {
        float* p = gstats + (size_t)(blockIdx.x * MB + tid) * 3;
        p[0] = ldsS[tid];
        p[1] = ldsP[tid];
        p[2] = ldsC[tid];
    }
}

// --- Kernel 3: combine the SPLITS partial stats per row -> per-row loss.
__global__ void row_loss(const float* __restrict__ gstats,
                         float* __restrict__ rowloss) {
    const int row = blockIdx.x * 256 + threadIdx.x;   // 0..8191
    const int strip = row >> 6;
    const int r = row & (MB - 1);
    float S = 0.f, P = 0.f, C = 0.f;
#pragma unroll
    for (int s = 0; s < SPLITS; ++s) {
        const float* p = gstats + (size_t)((strip * SPLITS + s) * MB + r) * 3;
        S += p[0]; P += p[1]; C += p[2];
    }
    const float L = __logf(S + 1e-8f);
    rowloss[row] = -(P - C * L) / (C + 1e-8f);        // T/baseT == 1
}

// --- Kernel 4: deterministic fixed-tree reduction of 8192 row losses.
__global__ void reduce_final(const float* __restrict__ rowloss,
                             float* __restrict__ out) {
    __shared__ float red[256];
    const int t = threadIdx.x;
    float s = 0.f;
#pragma unroll
    for (int k = 0; k < N_SZ / 256; ++k) s += rowloss[t + 256 * k];
    red[t] = s;
    __syncthreads();
    for (int w = 128; w >= 1; w >>= 1) {
        if (t < w) red[t] += red[t + w];
        __syncthreads();
    }
    if (t == 0) out[0] = red[0] * (1.f / (float)N_SZ);
}

extern "C" void kernel_launch(void* const* d_in, const int* in_sizes, int n_in,
                              void* d_out, int out_size, void* d_ws, size_t ws_size,
                              hipStream_t stream) {
    const float* feats  = (const float*)d_in[0];   // [B, V, D] f32
    const int*   labels = (const int*)d_in[1];     // [B] i32
    float*       out    = (float*)d_out;           // scalar f32

    // Workspace: [0, 8MB) bf16 features; then stats (512*64*3 f32 = 384 KB);
    // then per-row losses (32 KB).
    unsigned short* Fbf = (unsigned short*)d_ws;
    float* gstats  = (float*)((char*)d_ws + (size_t)N_SZ * D_SZ * sizeof(unsigned short));
    float* rowloss = gstats + (size_t)(N_SZ / MB) * SPLITS * MB * 3;

    normalize_bf16<<<N_SZ / 8, 256, 0, stream>>>(feats, Fbf);
    supcon_main<<<(N_SZ / MB) * SPLITS, 256, 0, stream>>>(Fbf, labels, gstats);
    row_loss<<<N_SZ / 256, 256, 0, stream>>>(gstats, rowloss);
    reduce_final<<<1, 256, 0, stream>>>(rowloss, out);
}
